// MAB_5446018531772
// MI455X (gfx1250) — matrix operand order repaired and checked by reading, not compiled, not run on hardware
//
#include <hip/hip_runtime.h>
#include <hip/hip_bf16.h>
#include <math.h>

typedef __attribute__((ext_vector_type(16))) __bf16 v16bf;
typedef __attribute__((ext_vector_type(8)))  __bf16 v8bf;
typedef __attribute__((ext_vector_type(8)))  float  v8f;

#define DEV static __device__ __forceinline__

DEV unsigned short f2bf(float f) {            // round-to-nearest-even f32 -> bf16
  unsigned u = __builtin_bit_cast(unsigned, f);
  u += 0x7fffu + ((u >> 16) & 1u);
  return (unsigned short)(u >> 16);
}
DEV unsigned pk2(float a, float b) {
  return (unsigned)f2bf(a) | ((unsigned)f2bf(b) << 16);
}
DEV v8f wmma_bf16(v16bf a, v16bf b, v8f c) {
  return __builtin_amdgcn_wmma_f32_16x16x32_bf16(false, a, false, b, (short)0, c, false, false);
}

// ---------------- Tensor Data Mover (async 2D bf16 tile copy global->LDS) ---
#if defined(__AMDGCN__) && defined(__has_builtin)
#if __has_builtin(__builtin_amdgcn_tensor_load_to_lds) && __has_builtin(__builtin_amdgcn_s_wait_tensorcnt)
#define USE_TDM 1
#endif
#endif
#ifndef USE_TDM
#define USE_TDM 0
#endif

#if USE_TDM
typedef __attribute__((ext_vector_type(4))) unsigned v4u;
typedef __attribute__((ext_vector_type(8))) int      v8i;
typedef __attribute__((ext_vector_type(4))) int      v4i;

// tile_w elems/row (bf16), tile_h rows, row stride = stride_elems.
// LDS row padding: pad_int_code (DWORDs before pad: 3->16,4->32), pad_amt_code
// (DWORDs of pad: 3->4) -> produces the +8-element padded LDS rows used below.
DEV void tdm_load_2d(unsigned lds_byte_off, const void* gaddr, unsigned tile_w,
                     unsigned tile_h, unsigned stride_elems,
                     unsigned pad_int_code, unsigned pad_amt_code) {
  const unsigned long long ga = (unsigned long long)(uintptr_t)gaddr;
  v4u g0;
  g0[0] = 1u;                                               // count=1 (user D#)
  g0[1] = lds_byte_off;                                     // LDS dest (bytes)
  g0[2] = (unsigned)(ga & 0xffffffffu);
  g0[3] = (unsigned)((ga >> 32) & 0x1ffffffu) | (2u << 30); // addr hi | type=2
  v8i g1;
  g1[0] = (int)((1u << 16) | (1u << 20) | (pad_int_code << 22) | (pad_amt_code << 25));
  g1[1] = (int)((tile_w & 0xffffu) << 16);                  // tensor_dim0 lo16
  g1[2] = (int)(((tile_w >> 16) & 0xffffu) | ((tile_h & 0xffffu) << 16));
  g1[3] = (int)(((tile_h >> 16) & 0xffffu) | ((tile_w & 0xffffu) << 16)); // tile_dim0
  g1[4] = (int)(tile_h & 0xffffu);                          // tile_dim1 (dim2=0)
  g1[5] = (int)stride_elems;                                // tensor_dim0_stride
  g1[6] = 0;
  g1[7] = 0;
  const v4i z4 = {0, 0, 0, 0};
#if __clang_major__ >= 23
  const v8i z8 = {0, 0, 0, 0, 0, 0, 0, 0};
  __builtin_amdgcn_tensor_load_to_lds(g0, g1, z4, z4, z8, 0);
#else
  __builtin_amdgcn_tensor_load_to_lds(g0, g1, z4, z4, 0);
#endif
}
#endif

// ---------------------------------------------------------------------------
// Elementwise f32 -> bf16 (8 elems/thread).
// ---------------------------------------------------------------------------
__global__ __launch_bounds__(256) void conv_bf16_kernel(
    const float* __restrict__ x, unsigned short* __restrict__ y, size_t n) {
  const size_t i = ((size_t)blockIdx.x * 256 + threadIdx.x) * 8;
  if (i + 8 <= n) {
    const float4 a = *(const float4*)(x + i);
    const float4 b = *(const float4*)(x + i + 4);
    uint4 o;
    o.x = pk2(a.x, a.y); o.y = pk2(a.z, a.w);
    o.z = pk2(b.x, b.y); o.w = pk2(b.z, b.w);
    *(uint4*)(y + i) = o;
  }
}

// ---------------------------------------------------------------------------
// W[K,N] f32 -> Wt[N,K] bf16  (32x32 LDS tiles, 32x8 threads)
// ---------------------------------------------------------------------------
__global__ __launch_bounds__(256) void transpose_bf16_kernel(
    const float* __restrict__ W, unsigned short* __restrict__ Wt, int K, int N) {
  __shared__ float t[32][33];
  const int tx = threadIdx.x & 31, ty = threadIdx.x >> 5;
  const int n0 = blockIdx.x * 32, k0 = blockIdx.y * 32;
#pragma unroll
  for (int j = 0; j < 4; ++j)
    t[ty + 8 * j][tx] = W[(size_t)(k0 + ty + 8 * j) * N + n0 + tx];
  __syncthreads();
#pragma unroll
  for (int j = 0; j < 4; ++j)
    Wt[(size_t)(n0 + ty + 8 * j) * K + k0 + tx] = f2bf(t[tx][ty + 8 * j]);
}

// ---------------------------------------------------------------------------
// GEMM: acc = A[M,K](bf16) @ Wt[N,K](bf16)^T + bias.  TDM double-buffered.
//   out  (f32, optional); fuse_gelu: out = res + gelu(acc+bias)
//   outb (bf16, optional): row-major * bscale, or transV per-head layout
//         outb[((b*H+h)*64+d)*SkV + key]
// ---------------------------------------------------------------------------
__global__ __launch_bounds__(256) void gemm_bf16_kernel(
    const unsigned short* __restrict__ A, const unsigned short* __restrict__ Wt,
    const float* __restrict__ bias, const float* __restrict__ res,
    float* __restrict__ out, unsigned short* __restrict__ outb,
    float bscale, int transV, int M, int N, int K, int SkV, int fuse_gelu)
{
  __shared__ __align__(16) unsigned short Asm[2][128 * 40];
  __shared__ __align__(16) unsigned short Bsm[2][128 * 40];

  const int tid  = threadIdx.x;
  const int lane = tid & 31;
  const int wid  = tid >> 5;
  const int waveM = wid >> 1;       // 0..3
  const int waveN = wid & 1;        // 0..1
  const int l16  = lane & 15;
  const int half = lane >> 4;

  const int mbase = blockIdx.y * 128;
  const int nbase = blockIdx.x * 128;
  const unsigned short* Arow = A  + (size_t)mbase * K;
  const unsigned short* Wrow = Wt + (size_t)nbase * K;

  v8f acc[2][4];
  const v8f zro = {0.f,0.f,0.f,0.f,0.f,0.f,0.f,0.f};
#pragma unroll
  for (int i = 0; i < 2; ++i)
#pragma unroll
    for (int j = 0; j < 4; ++j) acc[i][j] = zro;

  const int nk = K >> 5;

#if USE_TDM
  const unsigned a_off[2] = {(unsigned)(uintptr_t)&Asm[0][0], (unsigned)(uintptr_t)&Asm[1][0]};
  const unsigned b_off[2] = {(unsigned)(uintptr_t)&Bsm[0][0], (unsigned)(uintptr_t)&Bsm[1][0]};
  if (wid == 0) {   // prologue: tile 0 into buffer 0
    tdm_load_2d(a_off[0], Arow, 32, 128, (unsigned)K, 3, 3);
    tdm_load_2d(b_off[0], Wrow, 32, 128, (unsigned)K, 3, 3);
  }
#endif

  for (int it = 0; it < nk; ++it) {
#if USE_TDM
    if (wid == 0) {
      if (it + 1 < nk) {   // queue next tile, then wait for current pair
        tdm_load_2d(a_off[(it + 1) & 1], Arow + (it + 1) * 32, 32, 128, (unsigned)K, 3, 3);
        tdm_load_2d(b_off[(it + 1) & 1], Wrow + (it + 1) * 32, 32, 128, (unsigned)K, 3, 3);
        __builtin_amdgcn_s_wait_tensorcnt(2);
      } else {
        __builtin_amdgcn_s_wait_tensorcnt(0);
      }
    }
    __syncthreads();
    const unsigned short* As = Asm[it & 1];
    const unsigned short* Bs = Bsm[it & 1];
#else
#pragma unroll
    for (int i = 0; i < 2; ++i) {          // 128 rows x 4 chunks of 8 bf16
      const int idx = tid + 256 * i;
      const int row = idx >> 2, c8 = idx & 3;
      *(uint4*)(&Asm[0][row * 40 + c8 * 8]) = *(const uint4*)(Arow + (size_t)row * K + it * 32 + c8 * 8);
      *(uint4*)(&Bsm[0][row * 40 + c8 * 8]) = *(const uint4*)(Wrow + (size_t)row * K + it * 32 + c8 * 8);
    }
    __syncthreads();
    const unsigned short* As = Asm[0];
    const unsigned short* Bs = Bsm[0];
#endif

    v16bf afrag[2];
#pragma unroll
    for (int mf = 0; mf < 2; ++mf) {
      const int row = waveM * 32 + mf * 16 + l16;
      const v8bf r0 = *(const v8bf*)(&As[row * 40 + half * 16]);
      const v8bf r1 = *(const v8bf*)(&As[row * 40 + half * 16 + 8]);
#pragma unroll
      for (int j = 0; j < 8; ++j) { afrag[mf][j] = r0[j]; afrag[mf][j + 8] = r1[j]; }
    }
#pragma unroll
    for (int nf = 0; nf < 4; ++nf) {
      const int nrow = waveN * 64 + nf * 16 + l16;
      const v16bf bfrag = *(const v16bf*)(&Bs[nrow * 40 + half * 16]);
#pragma unroll
      for (int mf = 0; mf < 2; ++mf)
        acc[mf][nf] = wmma_bf16(afrag[mf], bfrag, acc[mf][nf]);
    }
    __syncthreads();
  }

  const int Hh = N >> 6;
#pragma unroll
  for (int nf = 0; nf < 4; ++nf) {
    const int col = nbase + waveN * 64 + nf * 16 + l16;
    const float bv = bias[col];
#pragma unroll
    for (int mf = 0; mf < 2; ++mf) {
#pragma unroll
      for (int r = 0; r < 8; ++r) {
        const int row = mbase + waveM * 32 + mf * 16 + r + half * 8;
        float v = acc[mf][nf][r] + bv;
        if (fuse_gelu) {
          v = 0.5f * v * (1.0f + erff(v * 0.70710678118654752f));
          v += res[(size_t)row * N + col];
        }
        if (out) out[(size_t)row * N + col] = v;
        if (outb) {
          if (transV) {
            const int bb = row / SkV, key = row % SkV;
            const int hh = col >> 6, d = col & 63;
            outb[((size_t)(bb * Hh + hh) * 64 + d) * SkV + key] = f2bf(v);
          } else {
            outb[(size_t)row * N + col] = f2bf(v * bscale);
          }
        }
      }
    }
  }
}

// ---------------------------------------------------------------------------
// Flash attention on bf16 tensors (Qb pre-scaled by 1/sqrt(dh)); TDM
// double-buffered K/V tiles.  Grid (Sq/64, H, B); 4 waves, 16 queries each.
// ---------------------------------------------------------------------------
__global__ __launch_bounds__(128) void attn_kernel(
    const unsigned short* __restrict__ Qb, const unsigned short* __restrict__ Kb,
    const unsigned short* __restrict__ Vtb, const float* __restrict__ Qp,
    float* __restrict__ O, int Sq, int Sk, int D, int H)
{
  __shared__ __align__(16) unsigned short Ksm[2][64 * 72];   // [key][d]
  __shared__ __align__(16) unsigned short Vtsm[2][64 * 72];  // [d][key]
  __shared__ __align__(16) unsigned short Psm[4][16 * 72];   // per-wave P

  const int tid = threadIdx.x, lane = tid & 31, wid = tid >> 5;
  const int l16 = lane & 15, half = lane >> 4;
  const int b = blockIdx.z, h = blockIdx.y;
  const int q0 = blockIdx.x * 64 + wid * 16;
  const v8f zro = {0.f,0.f,0.f,0.f,0.f,0.f,0.f,0.f};

  const unsigned short* kg = Kb + (size_t)(b * Sk) * D + h * 64;     // + key*D
  const unsigned short* vg = Vtb + (size_t)(b * H + h) * 64 * Sk;    // + d*Sk + key

  v16bf qf[2];
  {
    const unsigned short* qrow = Qb + (size_t)(b * Sq + q0 + l16) * D + h * 64;
#pragma unroll
    for (int kk = 0; kk < 2; ++kk) {
      const v8bf r0 = *(const v8bf*)(qrow + kk * 32 + half * 16);
      const v8bf r1 = *(const v8bf*)(qrow + kk * 32 + half * 16 + 8);
#pragma unroll
      for (int j = 0; j < 8; ++j) { qf[kk][j] = r0[j]; qf[kk][j + 8] = r1[j]; }
    }
  }

  float mi[8], li[8];
  v8f oacc[4];
#pragma unroll
  for (int r = 0; r < 8; ++r) { mi[r] = -1e30f; li[r] = 0.f; }
#pragma unroll
  for (int f = 0; f < 4; ++f) oacc[f] = zro;

  const int nkt = Sk >> 6;

#if USE_TDM
  const unsigned k_off[2] = {(unsigned)(uintptr_t)&Ksm[0][0], (unsigned)(uintptr_t)&Ksm[1][0]};
  const unsigned v_off[2] = {(unsigned)(uintptr_t)&Vtsm[0][0], (unsigned)(uintptr_t)&Vtsm[1][0]};
  if (wid == 0) {
    tdm_load_2d(k_off[0], kg, 64, 64, (unsigned)D, 4, 3);
    tdm_load_2d(v_off[0], vg, 64, 64, (unsigned)Sk, 4, 3);
  }
#endif

  for (int it = 0; it < nkt; ++it) {
#if USE_TDM
    if (wid == 0) {
      if (it + 1 < nkt) {
        tdm_load_2d(k_off[(it + 1) & 1], kg + (size_t)(it + 1) * 64 * D, 64, 64, (unsigned)D, 4, 3);
        tdm_load_2d(v_off[(it + 1) & 1], vg + (it + 1) * 64, 64, 64, (unsigned)Sk, 4, 3);
        __builtin_amdgcn_s_wait_tensorcnt(2);
      } else {
        __builtin_amdgcn_s_wait_tensorcnt(0);
      }
    }
    __syncthreads();
    const unsigned short* Ks = Ksm[it & 1];
    const unsigned short* Vs = Vtsm[it & 1];
#else
#pragma unroll
    for (int i = 0; i < 4; ++i) {          // 64 rows x 8 chunks of 8 bf16
      const int idx = tid + 128 * i;
      const int row = idx >> 3, c8 = idx & 7;
      *(uint4*)(&Ksm[0][row * 72 + c8 * 8]) = *(const uint4*)(kg + (size_t)(it * 64 + row) * D + c8 * 8);
      *(uint4*)(&Vtsm[0][row * 72 + c8 * 8]) = *(const uint4*)(vg + (size_t)row * Sk + it * 64 + c8 * 8);
    }
    __syncthreads();
    const unsigned short* Ks = Ksm[0];
    const unsigned short* Vs = Vtsm[0];
#endif

    v8f s[4];
#pragma unroll
    for (int f = 0; f < 4; ++f) {
      const int key = f * 16 + l16;
      const v16bf b0 = *(const v16bf*)(&Ks[key * 72 + half * 16]);
      const v16bf b1 = *(const v16bf*)(&Ks[key * 72 + 32 + half * 16]);
      s[f] = wmma_bf16(qf[0], b0, zro);
      s[f] = wmma_bf16(qf[1], b1, s[f]);
    }

#pragma unroll
    for (int r = 0; r < 8; ++r) {
      float t = fmaxf(fmaxf(s[0][r], s[1][r]), fmaxf(s[2][r], s[3][r]));
      t = fmaxf(t, __shfl_xor(t, 1));
      t = fmaxf(t, __shfl_xor(t, 2));
      t = fmaxf(t, __shfl_xor(t, 4));
      t = fmaxf(t, __shfl_xor(t, 8));
      const float mn = fmaxf(mi[r], t);
      const float al = __expf(mi[r] - mn);
      mi[r] = mn;
      li[r] *= al;
#pragma unroll
      for (int f = 0; f < 4; ++f) oacc[f][r] *= al;
    }

    float rs[8];
#pragma unroll
    for (int r = 0; r < 8; ++r) rs[r] = 0.f;
#pragma unroll
    for (int f = 0; f < 4; ++f) {
#pragma unroll
      for (int r = 0; r < 8; ++r) {
        const float p = __expf(s[f][r] - mi[r]);
        rs[r] += p;
        Psm[wid][(r + half * 8) * 72 + f * 16 + l16] = f2bf(p);
      }
    }
#pragma unroll
    for (int r = 0; r < 8; ++r) {
      float t = rs[r];
      t += __shfl_xor(t, 1);
      t += __shfl_xor(t, 2);
      t += __shfl_xor(t, 4);
      t += __shfl_xor(t, 8);
      li[r] += t;
    }

    const unsigned short* pw = &Psm[wid][0];
#pragma unroll
    for (int kk = 0; kk < 2; ++kk) {
      v16bf pa;
      const v8bf r0 = *(const v8bf*)(pw + l16 * 72 + kk * 32 + half * 16);
      const v8bf r1 = *(const v8bf*)(pw + l16 * 72 + kk * 32 + half * 16 + 8);
#pragma unroll
      for (int j = 0; j < 8; ++j) { pa[j] = r0[j]; pa[j + 8] = r1[j]; }
#pragma unroll
      for (int f = 0; f < 4; ++f) {
        const v16bf vb = *(const v16bf*)(&Vs[(f * 16 + l16) * 72 + kk * 32 + half * 16]);
        oacc[f] = wmma_bf16(pa, vb, oacc[f]);
      }
    }
    __syncthreads();
  }

#pragma unroll
  for (int r = 0; r < 8; ++r) li[r] = 1.0f / li[r];
#pragma unroll
  for (int f = 0; f < 4; ++f) {
    const int d = h * 64 + f * 16 + l16;
#pragma unroll
    for (int r = 0; r < 8; ++r) {
      const int row = q0 + r + half * 8;
      const size_t gi = (size_t)(b * Sq + row) * D + d;
      O[gi] = Qp[gi] + oacc[f][r] * li[r];
    }
  }
}

// ---------------------------------------------------------------------------
// LayerNorm (D=1024), optional bf16 copy of the output.
// ---------------------------------------------------------------------------
__global__ __launch_bounds__(256) void layernorm_kernel(
    const float* __restrict__ x, const float* __restrict__ gamma,
    const float* __restrict__ beta, float* __restrict__ y,
    unsigned short* __restrict__ yb, int D)
{
  __shared__ float sm1[8], sm2[8];
  const int row = blockIdx.x, tid = threadIdx.x;
  const int lane = tid & 31, wid = tid >> 5;
  const float4 v = *(const float4*)(x + (size_t)row * D + tid * 4);
  float s = v.x + v.y + v.z + v.w;
  float q = v.x * v.x + v.y * v.y + v.z * v.z + v.w * v.w;
#pragma unroll
  for (int off = 16; off >= 1; off >>= 1) {
    s += __shfl_xor(s, off);
    q += __shfl_xor(q, off);
  }
  if (lane == 0) { sm1[wid] = s; sm2[wid] = q; }
  __syncthreads();
  float ts = 0.f, tq = 0.f;
#pragma unroll
  for (int i = 0; i < 8; ++i) { ts += sm1[i]; tq += sm2[i]; }
  const float inv = 1.0f / (float)D;
  const float mu = ts * inv;
  const float var = tq * inv - mu * mu;
  const float rstd = rsqrtf(var + 1e-5f);
  const float4 g  = *(const float4*)(gamma + tid * 4);
  const float4 bb = *(const float4*)(beta + tid * 4);
  float4 o;
  o.x = (v.x - mu) * rstd * g.x + bb.x;
  o.y = (v.y - mu) * rstd * g.y + bb.y;
  o.z = (v.z - mu) * rstd * g.z + bb.z;
  o.w = (v.w - mu) * rstd * g.w + bb.w;
  *(float4*)(y + (size_t)row * D + tid * 4) = o;
  if (yb)
    *(uint2*)(yb + (size_t)row * D + tid * 4) = make_uint2(pk2(o.x, o.y), pk2(o.z, o.w));
}

// ---------------------------------------------------------------------------
extern "C" void kernel_launch(void* const* d_in, const int* in_sizes, int n_in,
                              void* d_out, int out_size, void* d_ws, size_t ws_size,
                              hipStream_t stream)
{
  const float* Q  = (const float*)d_in[0];
  const float* K  = (const float*)d_in[1];
  const float* Wq = (const float*)d_in[2];
  const float* bq = (const float*)d_in[3];
  const float* Wk = (const float*)d_in[4];
  const float* bk = (const float*)d_in[5];
  const float* Wv = (const float*)d_in[6];
  const float* bv = (const float*)d_in[7];
  const float* Wo = (const float*)d_in[8];
  const float* bo = (const float*)d_in[9];
  const float* g0 = (const float*)d_in[10];
  const float* b0 = (const float*)d_in[11];
  const float* g1 = (const float*)d_in[12];
  const float* b1 = (const float*)d_in[13];

  const int Bn = 2, S = 2048, D = 1024, H = 16;
  const int M = Bn * S;                       // 4096 rows
  const size_t PLANE = (size_t)M * D;         // 4M elements
  const size_t WSZ   = (size_t)D * D;         // 1M elements

  float* ws = (float*)d_ws;
  float* Qp = ws;                  // f32 Q projection (attn residual)
  float* Oa = ws + PLANE;          // f32 attention output
  float* X  = ws + 2 * PLANE;      // f32 LN0 output
  float* Y  = Qp;                  // reuse (Qp dead after attention)
  unsigned short* u   = (unsigned short*)(ws + 3 * PLANE);
  unsigned short* Qxb = u;                 // bf16 input Q
  unsigned short* Kxb = u + PLANE;         // bf16 input K
  unsigned short* Qb  = u + 2 * PLANE;     // bf16 Qp * 0.125
  unsigned short* Kb  = u + 3 * PLANE;     // bf16 Kp
  unsigned short* Vtb = u + 4 * PLANE;     // bf16 V, head-transposed
  unsigned short* Xb  = Qxb;               // reuse (Qxb dead after gemm1)
  unsigned short* Wqt = u + 5 * PLANE;     // bf16 W^T planes
  unsigned short* Wkt = Wqt + WSZ;
  unsigned short* Wvt = Wkt + WSZ;
  unsigned short* Wot = Wvt + WSZ;

  const dim3 gblk(256);
  const dim3 ggrid(D / 128, M / 128);
  const dim3 tgrid(D / 32, D / 32);

  conv_bf16_kernel<<<(unsigned)(PLANE / 2048), 256, 0, stream>>>(Q, Qxb, PLANE);
  conv_bf16_kernel<<<(unsigned)(PLANE / 2048), 256, 0, stream>>>(K, Kxb, PLANE);
  transpose_bf16_kernel<<<tgrid, 256, 0, stream>>>(Wq, Wqt, D, D);
  transpose_bf16_kernel<<<tgrid, 256, 0, stream>>>(Wk, Wkt, D, D);
  transpose_bf16_kernel<<<tgrid, 256, 0, stream>>>(Wv, Wvt, D, D);
  transpose_bf16_kernel<<<tgrid, 256, 0, stream>>>(Wo, Wot, D, D);

  gemm_bf16_kernel<<<ggrid, gblk, 0, stream>>>(Qxb, Wqt, bq, nullptr, Qp, Qb, 0.125f, 0, M, D, D, S, 0);
  gemm_bf16_kernel<<<ggrid, gblk, 0, stream>>>(Kxb, Wkt, bk, nullptr, nullptr, Kb, 1.0f, 0, M, D, D, S, 0);
  gemm_bf16_kernel<<<ggrid, gblk, 0, stream>>>(Kb,  Wvt, bv, nullptr, nullptr, Vtb, 1.0f, 1, M, D, D, S, 0);
  attn_kernel<<<dim3(S / 64, H, Bn), 128, 0, stream>>>(Qb, Kb, Vtb, Qp, Oa, S, S, D, H);
  layernorm_kernel<<<M, 256, 0, stream>>>(Oa, g0, b0, X, Xb, D);
  gemm_bf16_kernel<<<ggrid, gblk, 0, stream>>>(Xb, Wot, bo, X, Y, nullptr, 1.0f, 0, M, D, D, S, 1);
  layernorm_kernel<<<M, 256, 0, stream>>>(Y, g1, b1, (float*)d_out, nullptr, D);
}
